// HFScatterMoEGatedMLP_35742717837900
// MI455X (gfx1250) — matrix operand order repaired and checked by reading, not compiled, not run on hardware
//
#include <hip/hip_runtime.h>
#include <hip/hip_bf16.h>

typedef __attribute__((ext_vector_type(16))) __bf16       v16bf;
typedef __attribute__((ext_vector_type(8)))  float        v8f;
typedef __attribute__((ext_vector_type(4)))  unsigned int u32x4;
typedef __attribute__((ext_vector_type(8)))  int          i32x8;
typedef __attribute__((ext_vector_type(4)))  int          i32x4;

#define H_DIM 2048
#define I_DIM 1024
#define E_NUM 8
#define T_NUM 512
#define R_DIM 16
#define M_BLK 64                 // rows (tokens) per block
#define MB_TILES (T_NUM / M_BLK) // 8
#define A_STRIDE 36              // 32 data dwords + 4 TDM pad dwords per row

// ---------------------------------------------------------------------------
// TDM gather: async-load `nrows` rows (16-bit indices packed in pk[0..7]) of
// 32 f32 elements each from gsrc (row stride `stride` elems) into LDS at
// lds_byte_off, padding 4 dwords after every 32 (row stride 36 dwords).
// D# layout per CDNA5 ISA.
// ---------------------------------------------------------------------------
__device__ __forceinline__ void tdm_gather_rows(
    unsigned lds_byte_off, const float* gsrc, int nrows,
    unsigned dim0, unsigned dim1, unsigned stride, const int* pk) {
    unsigned long long ga = (unsigned long long)(size_t)gsrc;
    u32x4 g0;
    g0[0] = 1u | (1u << 31);                                  // count=1, gather_mode, 16b idx
    g0[1] = lds_byte_off;                                     // lds_addr
    g0[2] = (unsigned)ga;                                     // global_addr[31:0]
    g0[3] = (unsigned)((ga >> 32) & 0x01ffffffu) | (2u << 30);// addr[56:32], type=2
    i32x8 g1;
    g1[0] = (2 << 16) | (1 << 20) | (4 << 22) | (3 << 25);    // 4B elems, pad 4dw/32dw
    g1[1] = (int)((dim0 & 0xffffu) << 16);                    // tensor_dim0[15:0]
    g1[2] = (int)((dim0 >> 16) | ((dim1 & 0xffffu) << 16));   // dim0[31:16] | dim1[15:0]
    g1[3] = (int)((dim1 >> 16) | (32u << 16));                // dim1[31:16] | tile_dim0=32
    g1[4] = nrows & 0xffff;                                   // tile_dim1 = #indices
    g1[5] = (int)stride;                                      // tensor_dim0_stride lo
    g1[6] = 0;
    g1[7] = 0;
    i32x4 g2, g3;
    g2[0] = pk[0]; g2[1] = pk[1]; g2[2] = pk[2]; g2[3] = pk[3];
    g3[0] = pk[4]; g3[1] = pk[5]; g3[2] = pk[6]; g3[3] = pk[7];
    asm volatile("tensor_load_to_lds %0, %1, %2, %3"
                 :: "s"(g0), "s"(g1), "s"(g2), "s"(g3) : "memory");
}

// ---------------------------------------------------------------------------
__global__ void moe_init_kernel(int* __restrict__ cnt) {
    int i = threadIdx.x;
    if (i < E_NUM) cnt[i] = 0;
}

// weff[e][h] = gate_weight[e][h] + 2 * sum_r gate_lora_B[e][r]*gate_lora_A[r][h]
__global__ __launch_bounds__(256) void moe_weff_kernel(
    const float* __restrict__ gw, const float* __restrict__ la,
    const float* __restrict__ lb, float* __restrict__ weff) {
    int idx = blockIdx.x * 256 + threadIdx.x;
    int e = idx / H_DIM, h = idx % H_DIM;
    float s = 0.f;
#pragma unroll
    for (int r = 0; r < R_DIM; ++r)
        s += lb[e * R_DIM + r] * la[r * H_DIM + h];
    weff[idx] = gw[idx] + 2.0f * s;
}

// router: logits -> softmax -> top-2 -> normalized weights + expert lists.
__global__ __launch_bounds__(256) void moe_router_kernel(
    const float* __restrict__ x, const float* __restrict__ weff,
    float* __restrict__ rw, int* __restrict__ cnt,
    int* __restrict__ list_tok, int* __restrict__ list_pos) {
    int t = blockIdx.x;
    int tid = threadIdx.x, wave = tid >> 5, lane = tid & 31;
    __shared__ float partial[E_NUM][32];
    __shared__ float logits[E_NUM];
    const float* xr = x + (size_t)t * H_DIM;
    const float* wr = weff + (size_t)wave * H_DIM;
    float p = 0.f;
    for (int h = lane; h < H_DIM; h += 32) p += xr[h] * wr[h];
    partial[wave][lane] = p;
    __syncthreads();
    if (tid < E_NUM) {
        float s = 0.f;
        for (int l = 0; l < 32; ++l) s += partial[tid][l];
        logits[tid] = s;
    }
    __syncthreads();
    if (tid == 0) {
        float mx = logits[0];
        for (int e = 1; e < E_NUM; ++e) mx = fmaxf(mx, logits[e]);
        float pr[E_NUM]; float sum = 0.f;
        for (int e = 0; e < E_NUM; ++e) { pr[e] = __expf(logits[e] - mx); sum += pr[e]; }
        for (int e = 0; e < E_NUM; ++e) pr[e] /= sum;
        int i0 = 0;
        for (int e = 1; e < E_NUM; ++e) if (pr[e] > pr[i0]) i0 = e;
        int i1 = (i0 == 0) ? 1 : 0;
        for (int e = 0; e < E_NUM; ++e)
            if (e != i0 && pr[e] > pr[i1]) i1 = e;
        float s2 = pr[i0] + pr[i1];
        rw[t * 2 + 0] = pr[i0] / s2;
        rw[t * 2 + 1] = pr[i1] / s2;
        int s0 = atomicAdd(&cnt[i0], 1);
        list_tok[i0 * T_NUM + s0] = t;  list_pos[i0 * T_NUM + s0] = t * 2 + 0;
        int s1 = atomicAdd(&cnt[i1], 1);
        list_tok[i1 * T_NUM + s1] = t;  list_pos[i1 * T_NUM + s1] = t * 2 + 1;
    }
}

// LoRA-A projections: ter[pos][r] = dot(src_row, A[e][r][:len])
__global__ __launch_bounds__(256) void moe_terA_kernel(
    const float* __restrict__ src, const float* __restrict__ la,
    const int* __restrict__ cnt, const int* __restrict__ list_tok,
    const int* __restrict__ list_pos, float* __restrict__ ter,
    int len, int use_tok) {
    int e = blockIdx.x;
    int idx = blockIdx.y * 256 + threadIdx.x;
    int slot = idx >> 4, r = idx & 15;
    if (slot >= cnt[e]) return;
    int pos = list_pos[e * T_NUM + slot];
    int row = use_tok ? list_tok[e * T_NUM + slot] : pos;
    const float* sr = src + (size_t)row * len;
    const float* ar = la + ((size_t)e * R_DIM + r) * len;
    float s = 0.f;
    for (int i = 0; i < len; ++i) s += sr[i] * ar[i];
    ter[pos * R_DIM + r] = s;
}

// ---------------------------------------------------------------------------
// GUP: block = (expert e, M-block of 64 routed tokens, 128 cols of I).
// 8 waves each own a 16-wide column pair (gate col n, up col n+I).
// A tiles TDM-gathered to LDS (4 x 16-row gathers, async, TENSORcnt).
// K loop over H in 32-steps; 4 A sub-tiles reuse each B fragment pair
// (8 v_wmma per B load pair). Epilogue: LoRA delta + SiLU(gate)*up.
// ---------------------------------------------------------------------------
__global__ __launch_bounds__(256) void moe_gup_kernel(
    const float* __restrict__ x, const float* __restrict__ Wgup,
    const float* __restrict__ loraB, const float* __restrict__ terg,
    const int* __restrict__ cnt, const int* __restrict__ list_tok,
    const int* __restrict__ list_pos, float* __restrict__ h_buf) {
    int e = blockIdx.x / MB_TILES;
    int mb = blockIdx.x % MB_TILES;
    int c = cnt[e];
    int m0 = mb * M_BLK;
    if (m0 >= c) return;
    int valid = c - m0; if (valid > M_BLK) valid = M_BLK;
    int ntile = blockIdx.y;                     // 0..7 over I/128
    int tid = threadIdx.x, wave = tid >> 5, lane = tid & 31;
    int n = lane & 15, hi = lane >> 4;
    int n0 = ntile * 128 + wave * 16;

    __shared__ float Atile[M_BLK][A_STRIDE];    // TDM-padded rows
    __shared__ float terlds[M_BLK][R_DIM];
    __shared__ int   poslds[M_BLK];

    // packed 16-bit gather indices (block-uniform -> scalar)
    int idxpk[32];
#pragma unroll
    for (int pp = 0; pp < 32; ++pp) {
        int r0 = m0 + 2 * pp, r1 = r0 + 1;
        unsigned t0 = (r0 < c) ? (unsigned)list_tok[e * T_NUM + r0] : 0u;
        unsigned t1 = (r1 < c) ? (unsigned)list_tok[e * T_NUM + r1] : 0u;
        idxpk[pp] = (int)(t0 | (t1 << 16));
    }

    if (tid < M_BLK)
        poslds[tid] = (tid < valid) ? list_pos[e * T_NUM + m0 + tid] : 0;
    __syncthreads();
#pragma unroll
    for (int rep = 0; rep < 4; ++rep) {
        int idx = tid + rep * 256;
        terlds[idx >> 4][idx & 15] = terg[poslds[idx >> 4] * R_DIM + (idx & 15)];
    }

    unsigned lds_base = (unsigned)(size_t)&Atile[0][0];
    const float* Wg = Wgup + (size_t)e * H_DIM * (2 * I_DIM);
    v8f cg[4] = {}; v8f cu[4] = {};

    for (int kb = 0; kb < H_DIM; kb += 32) {
        __syncthreads();
        if (wave == 0) {
#pragma unroll
            for (int g = 0; g < 4; ++g) {
                int grows = valid - g * 16;
                if (grows > 0) {
                    if (grows > 16) grows = 16;
                    tdm_gather_rows(lds_base + g * 16 * A_STRIDE * 4,
                                    x + kb, grows, H_DIM, T_NUM, H_DIM,
                                    &idxpk[g * 8]);
                }
            }
        }
        __builtin_amdgcn_s_wait_tensorcnt(0);
        __syncthreads();

        v16bf bg, bu;
#pragma unroll
        for (int j = 0; j < 16; ++j) {
            int krow = kb + (j >> 3) * 16 + hi * 8 + (j & 7);  // ISA 16-bit B 32x16 layout
            bg[j] = (__bf16)Wg[(size_t)krow * (2 * I_DIM) + n0 + n];
            bu[j] = (__bf16)Wg[(size_t)krow * (2 * I_DIM) + n0 + n + I_DIM];
        }
#pragma unroll
        for (int s = 0; s < 4; ++s) {
            v16bf a;
            int mrow = s * 16 + (lane & 15);
#pragma unroll
            for (int i = 0; i < 16; ++i) {
                int k = (i >> 3) * 16 + hi * 8 + (i & 7);  // ISA 16-bit A 16x32 layout
                a[i] = (__bf16)Atile[mrow][k];
            }
            cg[s] = __builtin_amdgcn_wmma_f32_16x16x32_bf16(false, a, false, bg, (short)0, cg[s], false, false);
            cu[s] = __builtin_amdgcn_wmma_f32_16x16x32_bf16(false, a, false, bu, (short)0, cu[s], false, false);
        }
    }
    // LoRA-B delta + SiLU epilogue
    float blg[16], blu[16];
#pragma unroll
    for (int r = 0; r < R_DIM; ++r) {
        blg[r] = loraB[((size_t)e * 2 * I_DIM + (n0 + n)) * R_DIM + r];
        blu[r] = loraB[((size_t)e * 2 * I_DIM + (n0 + n + I_DIM)) * R_DIM + r];
    }
#pragma unroll
    for (int s = 0; s < 4; ++s) {
#pragma unroll
        for (int v = 0; v < 8; ++v) {
            int m = s * 16 + v + hi * 8;
            float dg = 0.f, du = 0.f;
#pragma unroll
            for (int r = 0; r < R_DIM; ++r) {
                float t = terlds[m][r];
                dg += t * blg[r]; du += t * blu[r];
            }
            float g = cg[s][v] + 2.0f * dg;
            float u = cu[s][v] + 2.0f * du;
            float hres = (g * (1.0f / (1.0f + __expf(-g)))) * u;
            if (m < valid)
                h_buf[(size_t)poslds[m] * I_DIM + n0 + n] = hres;
        }
    }
}

// ---------------------------------------------------------------------------
// DOWN: h[T_e, I] @ down_proj[e][I, H] + LoRA; same TDM/WMMA structure.
// ---------------------------------------------------------------------------
__global__ __launch_bounds__(256) void moe_down_kernel(
    const float* __restrict__ h_buf, const float* __restrict__ Wd_all,
    const float* __restrict__ loraB, const float* __restrict__ terd,
    const int* __restrict__ cnt, const int* __restrict__ list_pos,
    float* __restrict__ out_buf) {
    int e = blockIdx.x / MB_TILES;
    int mb = blockIdx.x % MB_TILES;
    int c = cnt[e];
    int m0 = mb * M_BLK;
    if (m0 >= c) return;
    int valid = c - m0; if (valid > M_BLK) valid = M_BLK;
    int ntile = blockIdx.y;                     // 0..15 over H/128
    int tid = threadIdx.x, wave = tid >> 5, lane = tid & 31;
    int n = lane & 15, hi = lane >> 4;
    int n0 = ntile * 128 + wave * 16;

    __shared__ float Atile[M_BLK][A_STRIDE];
    __shared__ float terlds[M_BLK][R_DIM];
    __shared__ int   poslds[M_BLK];

    int idxpk[32];
#pragma unroll
    for (int pp = 0; pp < 32; ++pp) {
        int r0 = m0 + 2 * pp, r1 = r0 + 1;
        unsigned p0 = (r0 < c) ? (unsigned)list_pos[e * T_NUM + r0] : 0u;
        unsigned p1 = (r1 < c) ? (unsigned)list_pos[e * T_NUM + r1] : 0u;
        idxpk[pp] = (int)(p0 | (p1 << 16));
    }

    if (tid < M_BLK)
        poslds[tid] = (tid < valid) ? list_pos[e * T_NUM + m0 + tid] : 0;
    __syncthreads();
#pragma unroll
    for (int rep = 0; rep < 4; ++rep) {
        int idx = tid + rep * 256;
        terlds[idx >> 4][idx & 15] = terd[poslds[idx >> 4] * R_DIM + (idx & 15)];
    }

    unsigned lds_base = (unsigned)(size_t)&Atile[0][0];
    const float* Wd = Wd_all + (size_t)e * I_DIM * H_DIM;
    v8f cc[4] = {};

    for (int kb = 0; kb < I_DIM; kb += 32) {
        __syncthreads();
        if (wave == 0) {
#pragma unroll
            for (int g = 0; g < 4; ++g) {
                int grows = valid - g * 16;
                if (grows > 0) {
                    if (grows > 16) grows = 16;
                    tdm_gather_rows(lds_base + g * 16 * A_STRIDE * 4,
                                    h_buf + kb, grows, I_DIM, 2 * T_NUM, I_DIM,
                                    &idxpk[g * 8]);
                }
            }
        }
        __builtin_amdgcn_s_wait_tensorcnt(0);
        __syncthreads();

        v16bf b;
#pragma unroll
        for (int j = 0; j < 16; ++j) {
            int krow = kb + (j >> 3) * 16 + hi * 8 + (j & 7);
            b[j] = (__bf16)Wd[(size_t)krow * H_DIM + n0 + n];
        }
#pragma unroll
        for (int s = 0; s < 4; ++s) {
            v16bf a;
            int mrow = s * 16 + (lane & 15);
#pragma unroll
            for (int i = 0; i < 16; ++i) {
                int k = (i >> 3) * 16 + hi * 8 + (i & 7);
                a[i] = (__bf16)Atile[mrow][k];
            }
            cc[s] = __builtin_amdgcn_wmma_f32_16x16x32_bf16(false, a, false, b, (short)0, cc[s], false, false);
        }
    }
    float bl[16];
#pragma unroll
    for (int r = 0; r < R_DIM; ++r)
        bl[r] = loraB[((size_t)e * H_DIM + (n0 + n)) * R_DIM + r];
#pragma unroll
    for (int s = 0; s < 4; ++s) {
#pragma unroll
        for (int v = 0; v < 8; ++v) {
            int m = s * 16 + v + hi * 8;
            float d = 0.f;
#pragma unroll
            for (int r = 0; r < R_DIM; ++r) d += terlds[m][r] * bl[r];
            if (m < valid)
                out_buf[(size_t)poslds[m] * H_DIM + n0 + n] = cc[s][v] + 2.0f * d;
        }
    }
}

// combine: y[t] = rw0 * out_buf[2t] + rw1 * out_buf[2t+1]
__global__ __launch_bounds__(256) void moe_combine_kernel(
    const float* __restrict__ out_buf, const float* __restrict__ rw,
    float* __restrict__ y) {
    int t = blockIdx.x;
    int h = blockIdx.y * 256 + threadIdx.x;
    float w0 = rw[t * 2 + 0], w1 = rw[t * 2 + 1];
    y[(size_t)t * H_DIM + h] =
        w0 * out_buf[(size_t)(t * 2 + 0) * H_DIM + h] +
        w1 * out_buf[(size_t)(t * 2 + 1) * H_DIM + h];
}

// ---------------------------------------------------------------------------
extern "C" void kernel_launch(void* const* d_in, const int* in_sizes, int n_in,
                              void* d_out, int out_size, void* d_ws, size_t ws_size,
                              hipStream_t stream) {
    const float* x     = (const float*)d_in[0];
    const float* gw    = (const float*)d_in[1];
    const float* gla   = (const float*)d_in[2];
    const float* glb   = (const float*)d_in[3];
    const float* wgup  = (const float*)d_in[4];
    const float* wdown = (const float*)d_in[5];
    const float* gupA  = (const float*)d_in[6];
    const float* gupB  = (const float*)d_in[7];
    const float* dwnA  = (const float*)d_in[8];
    const float* dwnB  = (const float*)d_in[9];
    float* y = (float*)d_out;

    char* ws = (char*)d_ws;
    size_t off = 0;
    auto take = [&](size_t bytes) { char* p = ws + off; off += (bytes + 255) & ~(size_t)255; return p; };
    float* weff     = (float*)take(E_NUM * H_DIM * 4);
    int*   cnt      = (int*)  take(E_NUM * 4);
    int*   list_tok = (int*)  take(E_NUM * T_NUM * 4);
    int*   list_pos = (int*)  take(E_NUM * T_NUM * 4);
    float* rw       = (float*)take(T_NUM * 2 * 4);
    float* terg     = (float*)take(T_NUM * 2 * R_DIM * 4);
    float* terd     = (float*)take(T_NUM * 2 * R_DIM * 4);
    float* h_buf    = (float*)take((size_t)T_NUM * 2 * I_DIM * 4);
    float* out_buf  = (float*)take((size_t)T_NUM * 2 * H_DIM * 4);
    (void)ws_size; (void)in_sizes; (void)n_in; (void)out_size;

    moe_init_kernel<<<1, 32, 0, stream>>>(cnt);
    moe_weff_kernel<<<(E_NUM * H_DIM) / 256, 256, 0, stream>>>(gw, gla, glb, weff);
    moe_router_kernel<<<T_NUM, 256, 0, stream>>>(x, weff, rw, cnt, list_tok, list_pos);
    moe_terA_kernel<<<dim3(E_NUM, (T_NUM * R_DIM) / 256), 256, 0, stream>>>(
        x, gupA, cnt, list_tok, list_pos, terg, H_DIM, 1);
    moe_gup_kernel<<<dim3(E_NUM * MB_TILES, I_DIM / 128), 256, 0, stream>>>(
        x, wgup, gupB, terg, cnt, list_tok, list_pos, h_buf);
    moe_terA_kernel<<<dim3(E_NUM, (T_NUM * R_DIM) / 256), 256, 0, stream>>>(
        h_buf, dwnA, cnt, list_tok, list_pos, terd, I_DIM, 0);
    moe_down_kernel<<<dim3(E_NUM * MB_TILES, H_DIM / 128), 256, 0, stream>>>(
        h_buf, wdown, dwnB, terd, cnt, list_pos, out_buf);
    moe_combine_kernel<<<dim3(T_NUM, H_DIM / 256), 256, 0, stream>>>(out_buf, rw, y);
}
